// SmolVLM2Attention_60773787238432
// MI455X (gfx1250) — matrix operand order repaired and checked by reading, not compiled, not run on hardware
//
#include <hip/hip_runtime.h>
#include <hip/hip_bf16.h>

// ---------------------------------------------------------------------------
// SmolVLM2-style GQA attention block for MI455X (gfx1250, wave32, WMMA + TDM).
//   hidden[2,1024,2048] f32 ; Wq[2048,2048] Wk/Wv[2048,512] Wo[2048,2048] f32
//   out[2,1024,2048] f32
// GEMMs: v_wmma_f32_16x16x32_bf16 with Tensor-Data-Mover tile staging into LDS
// (pad-insertion used for bank-conflict-free strides). Attention: flash-style,
// causal mask computed analytically.
// ---------------------------------------------------------------------------

typedef __attribute__((ext_vector_type(16))) __bf16 v16bf;
typedef __attribute__((ext_vector_type(8)))  float  v8f;
typedef __attribute__((ext_vector_type(4)))  unsigned uint4v;
typedef __attribute__((ext_vector_type(8)))  int      int8v;
typedef __attribute__((ext_vector_type(4)))  int      int4v;

__device__ __forceinline__ unsigned short f32_to_bf16_rne(float f) {
  union { float f; unsigned u; } x; x.f = f;
  unsigned u = x.u;
  unsigned r = u + 0x7FFFu + ((u >> 16) & 1u);   // round-to-nearest-even
  return (unsigned short)(r >> 16);
}
__device__ __forceinline__ __bf16 bits_as_bf16(unsigned short s) {
  __bf16 b; __builtin_memcpy(&b, &s, 2); return b;
}
__device__ __forceinline__ __bf16 cvt_bf16(float f) {
  return bits_as_bf16(f32_to_bf16_rne(f));
}
__device__ __forceinline__ __bf16 load_bf16(const float* p)          { return cvt_bf16(*p); }
__device__ __forceinline__ __bf16 load_bf16(const unsigned short* p) { return bits_as_bf16(*p); }

// ---------------------------------------------------------------------------
// Tensor Data Mover: 2D tile (tile_d0 x tile_d1 elements) from global memory
// (row stride = stride_elems) into LDS at lds_addr, with optional LDS padding
// of 1 DWORD every 2^(pad_interval_code+1) DWORDs. Descriptor per ISA ch.8.
// ---------------------------------------------------------------------------
__device__ __forceinline__ void tdm_load_2d(unsigned lds_addr, const void* gaddr,
                                            unsigned data_size_code,   // 0=1B,1=2B,2=4B,3=8B
                                            unsigned tile_d0, unsigned tile_d1,
                                            unsigned tensor_d0, unsigned tensor_d1,
                                            unsigned long long stride_elems,
                                            unsigned pad_interval_code,
                                            unsigned pad_amount_code,
                                            bool pad_en) {
  unsigned long long ga = (unsigned long long)gaddr;
  uint4v g0;
  g0.x = 1u;                                             // count=1, user descriptor
  g0.y = lds_addr;                                       // LDS byte address
  g0.z = (unsigned)(ga & 0xFFFFFFFFu);                   // global_addr[31:0]
  g0.w = (unsigned)((ga >> 32) & 0x1FFFFFFu) | (2u << 30); // addr[56:32] | type=2

  int8v g1;
  unsigned w0 = (data_size_code << 16)
              | (pad_en ? (1u << 20) : 0u)
              | (pad_interval_code << 22)
              | (pad_amount_code << 25);
  g1[0] = (int)w0;                                       // mask=0, ds, pad cfg
  g1[1] = (int)((tensor_d0 & 0xFFFFu) << 16);            // abar=0 | tensor_dim0[15:0]
  g1[2] = (int)(((tensor_d0 >> 16) & 0xFFFFu) | ((tensor_d1 & 0xFFFFu) << 16));
  g1[3] = (int)(((tensor_d1 >> 16) & 0xFFFFu) | ((tile_d0 & 0xFFFFu) << 16));
  g1[4] = (int)(tile_d1 & 0xFFFFu);                      // tile_dim1 | tile_dim2=0
  g1[5] = (int)(stride_elems & 0xFFFFFFFFull);           // dim0_stride[31:0]
  g1[6] = (int)((stride_elems >> 32) & 0xFFFFull);       // dim0_stride[47:32] | dim1_stride lo = 0
  g1[7] = 0;

  int4v z4 = {0, 0, 0, 0};
#if defined(__clang_major__) && (__clang_major__ >= 23)
  int8v z8 = {0, 0, 0, 0, 0, 0, 0, 0};
  __builtin_amdgcn_tensor_load_to_lds(g0, g1, z4, z4, z8, 0);
#else
  __builtin_amdgcn_tensor_load_to_lds(g0, g1, z4, z4, 0);
#endif
}

// ---------------------------------------------------------------------------
// Generic tiled GEMM:  C[M x N](f32, ldc, +col off) = A[M x K] * B[K x N](f32)
// A is f32 or bf16-bits (ushort). Block = 64(M) x 128(N), 8 waves; each wave
// owns a 64x16 column strip (4 accumulators). BLOCK_K = 32 = one WMMA K step.
// Tiles staged by the Tensor Data Mover (wave 0), padded rows via TDM pad.
// ---------------------------------------------------------------------------
template <typename AT>
__global__ __launch_bounds__(256)
void gemm_wmma_bf16(const AT* __restrict__ A, const float* __restrict__ B,
                    float* __restrict__ C, int M, int N, int K,
                    int lda, int ldb, int ldc, int c_col_off) {
  // padded LDS row strides produced by TDM pad-insertion (+1 DWORD per row)
  constexpr int      AS    = (sizeof(AT) == 4) ? 33 : 34;  // elements per A row
  constexpr unsigned ADSC  = (sizeof(AT) == 4) ? 2u : 1u;  // data_size code
  constexpr unsigned APADI = (sizeof(AT) == 4) ? 4u : 3u;  // 32 / 16 DWORD interval

  __shared__ AT    aT[64 * AS];     // 64 x 32 tile (+1 DWORD pad per row)
  __shared__ float bT[32 * 129];    // 32 x 128 tile (+1 DWORD pad per row)

  const int tid  = threadIdx.x;
  const int wave = tid >> 5;               // 0..7
  const int lane = tid & 31;
  const int half = lane >> 4;
  const int l16  = lane & 15;
  const int mblk = blockIdx.y * 64;
  const int nblk = blockIdx.x * 128;
  (void)N;

  const unsigned aLds = (unsigned)(unsigned long long)(void*)aT;
  const unsigned bLds = (unsigned)(unsigned long long)(void*)bT;

  v8f acc[4] = {};

  for (int k0 = 0; k0 < K; k0 += 32) {
    // ---- TDM staging: A 64x32, B 32x128 (wave 0 issues, all waves barrier) --
    if (wave == 0) {
      tdm_load_2d(aLds, A + (size_t)mblk * lda + k0, ADSC,
                  /*tile*/ 32u, 64u, /*tensor*/ (unsigned)lda, (unsigned)M,
                  (unsigned long long)lda, APADI, 0u, true);
      tdm_load_2d(bLds, B + (size_t)k0 * ldb + nblk, 2u,
                  /*tile*/ 128u, 32u, /*tensor*/ (unsigned)ldb, (unsigned)K,
                  (unsigned long long)ldb, 6u, 0u, true);
      __builtin_amdgcn_s_wait_tensorcnt(0);
    }
    __syncthreads();

    // ---- B fragment: K x N = 32 x 16; table k order (8*half + e, 16 + 8*half + e-8) ----
    v16bf bfrag;
    {
      union { v16bf v; __bf16 e[16]; } u;
#pragma unroll
      for (int e = 0; e < 16; ++e)
        u.e[e] = cvt_bf16(bT[((e < 8) ? (e + half * 8) : (e - 8 + 16 + half * 8)) * 129 + wave * 16 + l16]);
      bfrag = u.v;
    }
    // ---- 4 A fragments (16x32 each) + WMMA ----
#pragma unroll
    for (int mi = 0; mi < 4; ++mi) {
      union { v16bf v; __bf16 e[16]; } ua;
      const int m = mi * 16 + l16;
#pragma unroll
      for (int e = 0; e < 16; ++e) {
        int kk = (e < 8) ? (e + half * 8) : (e - 8 + 16 + half * 8);
        ua.e[e] = load_bf16(&aT[m * AS + kk]);
      }
      acc[mi] = __builtin_amdgcn_wmma_f32_16x16x32_bf16(
          false, ua.v, false, bfrag, (short)0, acc[mi], false, false);
    }
    __syncthreads();
  }

  // ---- epilogue: C/D layout n = lane&15, M = r + 8*half ----
#pragma unroll
  for (int mi = 0; mi < 4; ++mi) {
    union { v8f v; float e[8]; } uo; uo.v = acc[mi];
    const int n = nblk + wave * 16 + l16;
#pragma unroll
    for (int r = 0; r < 8; ++r) {
      int m = mblk + mi * 16 + r + half * 8;
      C[(size_t)m * ldc + c_col_off + n] = uo.e[r];
    }
  }
}

// ---------------------------------------------------------------------------
// RoPE (split-half / Llama style) + convert to bf16, head-major [b,h,s,64].
// block = 32 threads (one per rotation pair), grid = (S, heads, B).
// ---------------------------------------------------------------------------
__global__ __launch_bounds__(32)
void rope_convert_qk(const float* __restrict__ QKV, unsigned short* __restrict__ Out,
                     int n_heads, int col_off) {
  const int i = threadIdx.x;         // 0..31 pair index
  const int s = blockIdx.x;
  const int h = blockIdx.y;
  const int b = blockIdx.z;
  const float* base = QKV + ((size_t)(b * 1024 + s)) * 3072 + col_off + h * 64;
  float x1 = base[i];
  float x2 = base[i + 32];
  // inv_freq = 10000^(-2i/64)
  float inv_freq = __expf(-(float)(2 * i) * (1.0f / 64.0f) * 9.210340371976184f);
  float ang = (float)s * inv_freq;
  float sn, cs; __sincosf(ang, &sn, &cs);
  float o1 = x1 * cs - x2 * sn;
  float o2 = x2 * cs + x1 * sn;
  unsigned short* orow = Out + ((size_t)((b * n_heads + h) * 1024 + s)) * 64;
  orow[i]      = f32_to_bf16_rne(o1);
  orow[i + 32] = f32_to_bf16_rne(o2);
}

__global__ __launch_bounds__(64)
void convert_v(const float* __restrict__ QKV, unsigned short* __restrict__ Vb) {
  const int d = threadIdx.x;         // 0..63
  const int s = blockIdx.x;
  const int h = blockIdx.y;          // kv head
  const int b = blockIdx.z;
  float v = QKV[((size_t)(b * 1024 + s)) * 3072 + 2560 + h * 64 + d];
  Vb[((size_t)((b * 8 + h) * 1024 + s)) * 64 + d] = f32_to_bf16_rne(v);
}

// ---------------------------------------------------------------------------
// Flash attention: grid (qblock=16, head=32, batch=2), block = 128 (4 waves).
// Each wave owns 16 query rows; kv processed in blocks of 32 with online
// softmax. Scores and PV both run on bf16 WMMA. Output written bf16 into
// attn[b*S+q][h*64+d] for the final projection GEMM.
// ---------------------------------------------------------------------------
__global__ __launch_bounds__(128)
void flash_attn(const unsigned short* __restrict__ Qb,
                const unsigned short* __restrict__ Kb,
                const unsigned short* __restrict__ Vb,
                unsigned short* __restrict__ attn) {
  __shared__ unsigned short pT[4][16][36];   // per-wave P re-layout patch

  const int tid  = threadIdx.x;
  const int wave = tid >> 5;                 // 0..3
  const int lane = tid & 31;
  const int half = lane >> 4;
  const int l16  = lane & 15;
  const int qb = blockIdx.x;                 // 0..15
  const int h  = blockIdx.y;                 // 0..31
  const int b  = blockIdx.z;                 // 0..1
  const int kvh = h >> 2;                    // GQA: 4 q heads per kv head
  const int q0  = qb * 64 + wave * 16;       // wave's first query row

  const unsigned short* Qbase = Qb + ((size_t)(b * 32 + h)  * 1024) * 64;
  const unsigned short* Kbase = Kb + ((size_t)(b * 8 + kvh) * 1024) * 64;
  const unsigned short* Vbase = Vb + ((size_t)(b * 8 + kvh) * 1024) * 64;

  // Q fragments: contraction over d=64 -> 2 k-steps of 32
  v16bf qf[2];
#pragma unroll
  for (int f = 0; f < 2; ++f) {
    union { v16bf v; __bf16 e[16]; } u;
    const int m = q0 + l16;
#pragma unroll
    for (int e = 0; e < 16; ++e) {
      int kk = ((e < 8) ? (e + half * 8) : (e - 8 + 16 + half * 8)) + f * 32;
      u.e[e] = bits_as_bf16(Qbase[(size_t)m * 64 + kk]);
    }
    qf[f] = u.v;
  }

  v8f accO[4] = {};                          // 16 x 64 output accum
  float mrun[8], lrun[8];
#pragma unroll
  for (int r = 0; r < 8; ++r) { mrun[r] = -1e30f; lrun[r] = 0.0f; }

  const int nkb = qb * 2 + 2;                // causal kv coverage, blocks of 32
  for (int kb = 0; kb < nkb; ++kb) {
    const int kvbase = kb * 32;

    // ---- scores S = (Q K^T) : two 16x16 n-subtiles of kv ----
    v8f s[2];
#pragma unroll
    for (int j = 0; j < 2; ++j) {
      v8f t = {};
#pragma unroll
      for (int f = 0; f < 2; ++f) {
        union { v16bf v; __bf16 e[16]; } u;
        const int kvrow = kvbase + j * 16 + l16;   // B-layout: n = lane&15
#pragma unroll
        for (int e = 0; e < 16; ++e) {
          int d = ((e < 8) ? (e + half * 8) : (e - 8 + 16 + half * 8)) + f * 32;          // B-layout: table k order
          u.e[e] = bits_as_bf16(Kbase[(size_t)kvrow * 64 + d]);
        }
        t = __builtin_amdgcn_wmma_f32_16x16x32_bf16(
            false, qf[f], false, u.v, (short)0, t, false, false);
      }
      s[j] = t;
    }

    // ---- online softmax (rows live across the 16-lane N groups) ----
    union { v8f v; float e[8]; } u0, u1;
    u0.v = s[0]; u1.v = s[1];
#pragma unroll
    for (int r = 0; r < 8; ++r) {
      const int mg = q0 + r + half * 8;            // global query row
      const int c0 = kvbase + l16;
      const int c1 = kvbase + 16 + l16;
      float v0 = u0.e[r] * 0.125f;  if (c0 > mg) v0 = -1e9f;
      float v1 = u1.e[r] * 0.125f;  if (c1 > mg) v1 = -1e9f;
      float rowm = fmaxf(v0, v1);
#pragma unroll
      for (int off = 1; off < 16; off <<= 1)
        rowm = fmaxf(rowm, __shfl_xor(rowm, off, 16));
      float mnew  = fmaxf(mrun[r], rowm);
      float alpha = __expf(mrun[r] - mnew);
      float p0 = __expf(v0 - mnew);
      float p1 = __expf(v1 - mnew);
      float rs = p0 + p1;
#pragma unroll
      for (int off = 1; off < 16; off <<= 1)
        rs += __shfl_xor(rs, off, 16);
      lrun[r] = lrun[r] * alpha + rs;
      mrun[r] = mnew;
#pragma unroll
      for (int d = 0; d < 4; ++d) {
        union { v8f v; float e[8]; } uo; uo.v = accO[d];
        uo.e[r] *= alpha;
        accO[d] = uo.v;
      }
      // C-layout -> LDS (bf16) for A-layout reload
      pT[wave][r + half * 8][l16]      = f32_to_bf16_rne(p0);
      pT[wave][r + half * 8][16 + l16] = f32_to_bf16_rne(p1);
    }
    __syncthreads();

    // ---- reload P as an A fragment (16 x 32 over kv) ----
    v16bf pfrag;
    {
      union { v16bf v; __bf16 e[16]; } u;
#pragma unroll
      for (int e = 0; e < 16; ++e) {
        int kk = (e < 8) ? (e + half * 8) : (e - 8 + 16 + half * 8);
        u.e[e] = bits_as_bf16(pT[wave][l16][kk]);
      }
      pfrag = u.v;
    }

    // ---- O += P * V : 4 head-dim subtiles of 16 ----
#pragma unroll
    for (int d = 0; d < 4; ++d) {
      union { v16bf v; __bf16 e[16]; } u;
#pragma unroll
      for (int e = 0; e < 16; ++e) {
        int kv = kvbase + ((e < 8) ? (e + half * 8) : (e - 8 + 16 + half * 8));            // B-layout k
        u.e[e] = bits_as_bf16(Vbase[(size_t)kv * 64 + d * 16 + l16]);
      }
      accO[d] = __builtin_amdgcn_wmma_f32_16x16x32_bf16(
          false, pfrag, false, u.v, (short)0, accO[d], false, false);
    }
    __syncthreads();
  }

  // ---- epilogue: normalize and store bf16 into attn[b*S+q][h*64+d] ----
#pragma unroll
  for (int d = 0; d < 4; ++d) {
    union { v8f v; float e[8]; } uo; uo.v = accO[d];
#pragma unroll
    for (int r = 0; r < 8; ++r) {
      const int qg = q0 + r + half * 8;
      float val = uo.e[r] / lrun[r];
      attn[((size_t)(b * 1024 + qg)) * 2048 + h * 64 + d * 16 + l16] =
          f32_to_bf16_rne(val);
    }
  }
}

// ---------------------------------------------------------------------------
// Host-side orchestration (graph-capture safe: only kernel launches).
// Workspace layout (bytes):
//   [0,            25165824)  QKV f32      [2048 x 3072]
//   [25165824,     33554432)  Q  bf16      [2,32,1024,64]
//   [33554432,     35651584)  K  bf16      [2, 8,1024,64]
//   [35651584,     37748736)  V  bf16      [2, 8,1024,64]
//   [37748736,     46137344)  attn bf16    [2048 x 2048]
// ---------------------------------------------------------------------------
extern "C" void kernel_launch(void* const* d_in, const int* in_sizes, int n_in,
                              void* d_out, int out_size, void* d_ws, size_t ws_size,
                              hipStream_t stream) {
  (void)in_sizes; (void)n_in; (void)out_size; (void)ws_size;
  const float* hidden = (const float*)d_in[0];
  // d_in[1] = attention_mask (causal) -- computed analytically, unused
  const float* Wq = (const float*)d_in[2];
  const float* Wk = (const float*)d_in[3];
  const float* Wv = (const float*)d_in[4];
  const float* Wo = (const float*)d_in[5];
  float* out = (float*)d_out;

  char* ws = (char*)d_ws;
  float*          qkv   = (float*)ws;
  unsigned short* Qb    = (unsigned short*)(ws + 25165824);
  unsigned short* Kbf   = (unsigned short*)(ws + 33554432);
  unsigned short* Vbf   = (unsigned short*)(ws + 35651584);
  unsigned short* attnb = (unsigned short*)(ws + 37748736);

  // 1) QKV projections (f32 A, f32 B -> bf16 WMMA -> f32 C, ldc = 3072)
  gemm_wmma_bf16<float><<<dim3(2048 / 128, 2048 / 64), 256, 0, stream>>>(
      hidden, Wq, qkv, 2048, 2048, 2048, 2048, 2048, 3072, 0);
  gemm_wmma_bf16<float><<<dim3(512 / 128, 2048 / 64), 256, 0, stream>>>(
      hidden, Wk, qkv, 2048, 512, 2048, 2048, 512, 3072, 2048);
  gemm_wmma_bf16<float><<<dim3(512 / 128, 2048 / 64), 256, 0, stream>>>(
      hidden, Wv, qkv, 2048, 512, 2048, 2048, 512, 3072, 2560);

  // 2) RoPE + bf16 conversion into head-major layouts
  rope_convert_qk<<<dim3(1024, 32, 2), 32, 0, stream>>>(qkv, Qb, 32, 0);
  rope_convert_qk<<<dim3(1024, 8, 2), 32, 0, stream>>>(qkv, Kbf, 8, 2048);
  convert_v<<<dim3(1024, 8, 2), 64, 0, stream>>>(qkv, Vbf);

  // 3) Flash attention (causal, GQA 4:1) -> attn bf16 [b*S][h*64+d]
  flash_attn<<<dim3(16, 32, 2), 128, 0, stream>>>(Qb, Kbf, Vbf, attnb);

  // 4) Output projection: attn(bf16) @ Wo(f32->bf16) -> out f32
  gemm_wmma_bf16<unsigned short><<<dim3(2048 / 128, 2048 / 64), 256, 0, stream>>>(
      attnb, Wo, out, 2048, 2048, 2048, 2048, 2048, 2048, 0);
}
